// FusedAttention_31645319037119
// MI455X (gfx1250) — compile-verified, matrix operand order checked by reading, not run on hardware
//
#include <hip/hip_runtime.h>

typedef __attribute__((ext_vector_type(16))) __bf16 v16bf;
typedef __attribute__((ext_vector_type(8)))  float  v8f;
typedef unsigned short u16;
typedef unsigned int   u32;

// Problem constants (static per reference setup_inputs)
constexpr int BATCH = 4;
constexpr int SEQ   = 2177;      // K + 1 + T
constexpr int SEQP  = 2184;      // padded (multiple of 8) row stride for V^T
constexpr int CDIM  = 768;
constexpr int NH    = 12;
constexpr int DH    = 64;
constexpr int KMEM  = 128;
constexpr int MROWS = BATCH * SEQ;      // 8708
constexpr int N_QKV = 3 * CDIM;         // 2304
constexpr int NQT   = (SEQ + 15) / 16;  // 137

constexpr int BM = 128, BN = 128, BKT = 32;

union Frag16 { v16bf v; u16 u[16]; uint4 q2[2]; };

__device__ __forceinline__ u16 f2bf(float f) {
  u32 x = __float_as_uint(f);
  x = (x + 0x7FFFu + ((x >> 16) & 1u)) >> 16;   // round-to-nearest-even
  return (u16)x;
}
__device__ __forceinline__ u32 pack2(float lo, float hi) {
  return (u32)f2bf(lo) | ((u32)f2bf(hi) << 16);
}
__device__ __forceinline__ int imin(int a, int b) { return a < b ? a : b; }
__device__ __forceinline__ int imax(int a, int b) { return a > b ? a : b; }

// ---------------------------------------------------------------------------
// Prep 1: x (f32) -> xb (bf16), vectorized.
// ---------------------------------------------------------------------------
__global__ __launch_bounds__(256) void cvt_x_kernel(
    const float* __restrict__ x, u16* __restrict__ xb)
{
  const size_t total = (size_t)MROWS * CDIM;
  const size_t i = ((size_t)blockIdx.x * 256 + threadIdx.x) * 8;
  if (i >= total) return;
  float4 a = *(const float4*)(x + i);
  float4 b = *(const float4*)(x + i + 4);
  uint4 o;
  o.x = pack2(a.x, a.y); o.y = pack2(a.z, a.w);
  o.z = pack2(b.x, b.y); o.w = pack2(b.z, b.w);
  *(uint4*)(xb + i) = o;
}

// ---------------------------------------------------------------------------
// Prep 2: W (K x N, f32, row-major) -> Wt (N x K, bf16), LDS-tiled transpose.
// Grid: (N/32, K/32), block (32, 8).
// ---------------------------------------------------------------------------
__global__ __launch_bounds__(256) void transpose_w_kernel(
    const float* __restrict__ W, u16* __restrict__ Wt, int K, int N)
{
  __shared__ u16 tile[32][33];
  const int tx = threadIdx.x, ty = threadIdx.y;
  const int n0 = blockIdx.x * 32, k0 = blockIdx.y * 32;
  #pragma unroll
  for (int r = 0; r < 4; r++) {
    const int kl = ty + r * 8;
    tile[kl][tx] = f2bf(W[(size_t)(k0 + kl) * N + n0 + tx]);   // coalesced read
  }
  __syncthreads();
  #pragma unroll
  for (int r = 0; r < 4; r++) {
    const int nl = ty + r * 8;
    Wt[(size_t)(n0 + nl) * K + k0 + tx] = tile[tx][nl];        // coalesced write
  }
}

// ---------------------------------------------------------------------------
// Fused GEMM: C = A(bf16, M x 768) @ Bt^T (Bt is bf16 [n][k]).
// MODE 0: N=2304, scatter into q (x1/8), k (B,H,L,d) and v^T (B,H,d,SEQP).
// MODE 1: N=768, write f32 out (B*L, C).
// ---------------------------------------------------------------------------
template <int MODE>
__global__ __launch_bounds__(256) void gemm_kernel(
    const u16* __restrict__ A, const u16* __restrict__ Bt,
    u16* __restrict__ qb, u16* __restrict__ kb, u16* __restrict__ vt,
    float* __restrict__ out)
{
  __shared__ __align__(16) u16 sA[BM * BKT];   // [m][k]
  __shared__ __align__(16) u16 sB[BN * BKT];   // [n][k]

  const int tid = threadIdx.x;
  const int lane = tid & 31, wid = tid >> 5;
  const int col16 = lane & 15, kg = lane >> 4;
  const int tm = blockIdx.x * BM, tn = blockIdx.y * BN;
  const int wm = (wid & 3) * 32, wn = (wid >> 2) * 64;

  v8f acc[2][4];
  #pragma unroll
  for (int i = 0; i < 2; i++)
    #pragma unroll
    for (int j = 0; j < 4; j++)
      #pragma unroll
      for (int r = 0; r < 8; r++) acc[i][j][r] = 0.0f;

  // staging: 256 threads, each copies 16 bf16 per tile (two b128 each way)
  const int srow = tid >> 1, sc0 = (tid & 1) * 16;
  const int agm = imin(tm + srow, MROWS - 1);            // clamp: no zero-fill
  const u16* aBase = A  + (size_t)agm * CDIM + sc0;
  const u16* bBase = Bt + (size_t)(tn + srow) * CDIM + sc0;

  for (int k0 = 0; k0 < CDIM; k0 += BKT) {
    const u16* ap = aBase + k0;
    const u16* bp = bBase + k0;
    *(uint4*)&sA[srow * BKT + sc0]     = *(const uint4*)ap;
    *(uint4*)&sA[srow * BKT + sc0 + 8] = *(const uint4*)(ap + 8);
    *(uint4*)&sB[srow * BKT + sc0]     = *(const uint4*)bp;
    *(uint4*)&sB[srow * BKT + sc0 + 8] = *(const uint4*)(bp + 8);
    if (k0 + BKT < CDIM) {
      __builtin_prefetch(ap + BKT, 0, 0);
      __builtin_prefetch(bp + BKT, 0, 0);
    }
    __syncthreads();

    Frag16 aF[2], bF[4];
    #pragma unroll
    for (int mi = 0; mi < 2; mi++) {
      const u16* p = &sA[(wm + mi * 16 + col16) * BKT + kg * 8];
      aF[mi].q2[0] = *(const uint4*)p;
      aF[mi].q2[1] = *(const uint4*)(p + 16);
    }
    #pragma unroll
    for (int ni = 0; ni < 4; ni++) {
      const u16* p = &sB[(wn + ni * 16 + col16) * BKT + kg * 8];
      bF[ni].q2[0] = *(const uint4*)p;
      bF[ni].q2[1] = *(const uint4*)(p + 16);
    }
    #pragma unroll
    for (int mi = 0; mi < 2; mi++)
      #pragma unroll
      for (int ni = 0; ni < 4; ni++)
        acc[mi][ni] = __builtin_amdgcn_wmma_f32_16x16x32_bf16(
            false, aF[mi].v, false, bF[ni].v, (short)0, acc[mi][ni], false, false);
    __syncthreads();
  }

  if (MODE == 0) {
    const int which = tn / CDIM;          // uniform: 128 | 768
    const int cbase = tn - which * CDIM;
    #pragma unroll
    for (int mi = 0; mi < 2; mi++) {
      #pragma unroll
      for (int r = 0; r < 8; r++) {
        const int grow = tm + wm + mi * 16 + ((lane < 16) ? r : 8 + r);
        if (grow >= MROWS) continue;
        const int bidx = grow / SEQ, l = grow - bidx * SEQ;
        #pragma unroll
        for (int ni = 0; ni < 4; ni++) {
          const int c = cbase + wn + ni * 16 + col16;
          const int hh = c >> 6, dd = c & 63;
          const float val = acc[mi][ni][r];
          if (which == 0) {
            qb[(((size_t)bidx * NH + hh) * SEQ + l) * DH + dd] = f2bf(val * 0.125f);
          } else if (which == 1) {
            kb[(((size_t)bidx * NH + hh) * SEQ + l) * DH + dd] = f2bf(val);
          } else {
            vt[(((size_t)bidx * NH + hh) * DH + dd) * SEQP + l] = f2bf(val);
          }
        }
      }
    }
  } else {
    #pragma unroll
    for (int mi = 0; mi < 2; mi++) {
      #pragma unroll
      for (int r = 0; r < 8; r++) {
        const int grow = tm + wm + mi * 16 + ((lane < 16) ? r : 8 + r);
        if (grow >= MROWS) continue;
        #pragma unroll
        for (int ni = 0; ni < 4; ni++)
          out[(size_t)grow * CDIM + tn + wn + ni * 16 + col16] = acc[mi][ni][r];
      }
    }
  }
}

// ---------------------------------------------------------------------------
// Flash attention helpers.  sP is a 16 x 64 bf16 LDS tile (row stride 64).
// Per-row key limit: lim(qi) = (qi == KMEM) ? SEQ-1 : max(qi, KMEM).
// (memory rows see [mem|state]; state row sees ALL; token rows causal.)
// ---------------------------------------------------------------------------

// Full 64-key tile: every key allowed for every row; no clamps, no compares.
__device__ __forceinline__ void attn_tile64(
    const int kb0, const int q0, const int rowbase,
    const u16* __restrict__ kp, const u16* __restrict__ vtp,
    u16* __restrict__ sP, const Frag16 (&qF)[2], const v16bf onesv,
    float (&mI)[8], v8f (&o)[4], v8f &osum,
    const int col16, const int kg)
{
  float s[4][8];
  #pragma unroll
  for (int klo = 0; klo < 4; ++klo) {
    const int kidx = kb0 + klo * 16 + col16;
    v8f sf;
    #pragma unroll
    for (int r = 0; r < 8; r++) sf[r] = 0.0f;
    #pragma unroll
    for (int hd = 0; hd < 2; hd++) {
      Frag16 kf;
      const u16* p = kp + (size_t)kidx * DH + hd * 32 + kg * 8;
      kf.q2[0] = *(const uint4*)p;
      kf.q2[1] = *(const uint4*)(p + 16);
      sf = __builtin_amdgcn_wmma_f32_16x16x32_bf16(
          false, qF[hd].v, false, kf.v, (short)0, sf, false, false);
    }
    #pragma unroll
    for (int r = 0; r < 8; r++) s[klo][r] = sf[r];
  }
  // one softmax update per 64 keys
  #pragma unroll
  for (int r = 0; r < 8; r++) {
    float tmax = fmaxf(fmaxf(s[0][r], s[1][r]), fmaxf(s[2][r], s[3][r]));
    #pragma unroll
    for (int off = 8; off >= 1; off >>= 1)
      tmax = fmaxf(tmax, __shfl_xor(tmax, off, 16));
    const float mNew = fmaxf(mI[r], tmax);
    const float alpha = __expf(mI[r] - mNew);
    mI[r] = mNew;
    const int rowl = rowbase - q0 + r;
    #pragma unroll
    for (int klo = 0; klo < 4; ++klo)
      sP[rowl * 64 + klo * 16 + col16] = f2bf(__expf(s[klo][r] - mNew));
    #pragma unroll
    for (int c = 0; c < 4; c++) o[c][r] *= alpha;
    osum[r] *= alpha;
  }
  __syncthreads();
  #pragma unroll
  for (int kc = 0; kc < 2; kc++) {
    Frag16 pA;
    const u16* pp = &sP[col16 * 64 + kc * 32 + kg * 8];
    pA.q2[0] = *(const uint4*)pp;
    pA.q2[1] = *(const uint4*)(pp + 16);
    #pragma unroll
    for (int c = 0; c < 4; c++) {
      Frag16 vB;
      const u16* p = vtp + (size_t)(c * 16 + col16) * SEQP + kb0 + kc * 32 + kg * 8;
      vB.q2[0] = *(const uint4*)p;
      vB.q2[1] = *(const uint4*)(p + 16);
      o[c] = __builtin_amdgcn_wmma_f32_16x16x32_bf16(
          false, pA.v, false, vB.v, (short)0, o[c], false, false);
    }
    osum = __builtin_amdgcn_wmma_f32_16x16x32_bf16(
        false, pA.v, false, onesv, (short)0, osum, false, false);
  }
  __syncthreads();
}

// Masked 32-key tile (diagonal / tail / state-row scan).
__device__ __forceinline__ void attn_tile32_masked(
    const int kb0, const int q0, const int kend, const int rowbase,
    const u16* __restrict__ kp, const u16* __restrict__ vtp,
    u16* __restrict__ sP, const Frag16 (&qF)[2], const v16bf onesv,
    float (&mI)[8], v8f (&o)[4], v8f &osum,
    const int col16, const int kg)
{
  float s[2][8];
  #pragma unroll
  for (int klo = 0; klo < 2; ++klo) {
    const int key16 = kb0 + klo * 16;
    const int kidx = imin(key16 + col16, SEQ - 1);
    v8f sf;
    #pragma unroll
    for (int r = 0; r < 8; r++) sf[r] = 0.0f;
    #pragma unroll
    for (int hd = 0; hd < 2; hd++) {
      Frag16 kf;
      const u16* p = kp + (size_t)kidx * DH + hd * 32 + kg * 8;
      kf.q2[0] = *(const uint4*)p;
      kf.q2[1] = *(const uint4*)(p + 16);
      sf = __builtin_amdgcn_wmma_f32_16x16x32_bf16(
          false, qF[hd].v, false, kf.v, (short)0, sf, false, false);
    }
    const int ki = key16 + col16;
    #pragma unroll
    for (int r = 0; r < 8; r++) {
      const int row = rowbase + r;
      const int lim = (row == KMEM) ? (SEQ - 1) : imax(row, KMEM);  // state row sees all
      s[klo][r] = (ki <= lim && ki <= kend) ? sf[r] : -1e30f;
    }
  }
  #pragma unroll
  for (int r = 0; r < 8; r++) {
    float tmax = fmaxf(s[0][r], s[1][r]);
    #pragma unroll
    for (int off = 8; off >= 1; off >>= 1)
      tmax = fmaxf(tmax, __shfl_xor(tmax, off, 16));
    const float mNew = fmaxf(mI[r], tmax);
    const float alpha = __expf(mI[r] - mNew);
    mI[r] = mNew;
    const int rowl = rowbase - q0 + r;
    sP[rowl * 64 + col16]      = f2bf(__expf(s[0][r] - mNew));
    sP[rowl * 64 + 16 + col16] = f2bf(__expf(s[1][r] - mNew));
    #pragma unroll
    for (int c = 0; c < 4; c++) o[c][r] *= alpha;
    osum[r] *= alpha;
  }
  __syncthreads();
  Frag16 pA;
  {
    const u16* pp = &sP[col16 * 64 + kg * 8];
    pA.q2[0] = *(const uint4*)pp;
    pA.q2[1] = *(const uint4*)(pp + 16);
  }
  #pragma unroll
  for (int c = 0; c < 4; c++) {
    Frag16 vB;
    const u16* p = vtp + (size_t)(c * 16 + col16) * SEQP + kb0 + kg * 8;
    vB.q2[0] = *(const uint4*)p;
    vB.q2[1] = *(const uint4*)(p + 16);
    o[c] = __builtin_amdgcn_wmma_f32_16x16x32_bf16(
        false, pA.v, false, vB.v, (short)0, o[c], false, false);
  }
  osum = __builtin_amdgcn_wmma_f32_16x16x32_bf16(
      false, pA.v, false, onesv, (short)0, osum, false, false);
  __syncthreads();
}

// ---------------------------------------------------------------------------
// Flash attention, one wave per (b, h, 16-row query tile).
// ---------------------------------------------------------------------------
__global__ __launch_bounds__(32) void attn_kernel(
    const u16* __restrict__ q, const u16* __restrict__ k,
    const u16* __restrict__ vt, u16* __restrict__ y)
{
  __shared__ __align__(16) u16 sP[16 * 64];
  const int lane = threadIdx.x;
  const int col16 = lane & 15, kg = lane >> 4;
  const int qt = blockIdx.x % NQT;
  const int bh = blockIdx.x / NQT;
  const int hh = bh % NH, b = bh / NH;
  const u16* qp  = q  + (size_t)(b * NH + hh) * SEQ * DH;
  const u16* kp  = k  + (size_t)(b * NH + hh) * SEQ * DH;
  const u16* vtp = vt + (size_t)(b * NH + hh) * DH * SEQP;
  const int q0 = qt * 16;
  const int rowbase = q0 + ((lane < 16) ? 0 : 8);

  // Q fragments (A layout), two k=32 halves of d=64
  Frag16 qF[2];
  {
    const int qrow = imin(q0 + col16, SEQ - 1);
    #pragma unroll
    for (int hd = 0; hd < 2; hd++) {
      const u16* p = qp + (size_t)qrow * DH + hd * 32 + kg * 8;
      qF[hd].q2[0] = *(const uint4*)p;
      qF[hd].q2[1] = *(const uint4*)(p + 16);
    }
  }
  Frag16 ones;
  #pragma unroll
  for (int j = 0; j < 16; j++) ones.u[j] = 0x3F80;   // bf16(1.0)

  float mI[8];
  v8f o[4], osum;
  #pragma unroll
  for (int r = 0; r < 8; r++) { mI[r] = -1e30f; osum[r] = 0.0f; }
  #pragma unroll
  for (int c = 0; c < 4; c++)
    #pragma unroll
    for (int r = 0; r < 8; r++) o[c][r] = 0.0f;

  // Key range this tile needs:
  //  - tile containing the state row (q0 <= KMEM <= q0+15): all keys
  //  - otherwise: up to max(q0+15, KMEM), clamped
  const bool has_state = (q0 <= KMEM) && (KMEM <= q0 + 15);
  const int kend = has_state ? (SEQ - 1) : imin(imax(q0 + 15, KMEM), SEQ - 1);
  const int limit_min = imax(q0, KMEM);       // min over rows of allowed limit
  const int nfull64 = (limit_min + 1) / 64;   // 64-key tiles with kb0+63 <= limit_min
  const int ntiles32 = kend / 32 + 1;

  for (int t = 0; t < nfull64; ++t)
    attn_tile64(t * 64, q0, rowbase, kp, vtp, sP, qF, ones.v,
                mI, o, osum, col16, kg);
  for (int t = 2 * nfull64; t < ntiles32; ++t)
    attn_tile32_masked(t * 32, q0, kend, rowbase, kp, vtp, sP, qF, ones.v,
                       mI, o, osum, col16, kg);

  // epilogue: normalize by osum (row sums replicated across columns)
  #pragma unroll
  for (int r = 0; r < 8; r++) {
    const float inv = 1.0f / osum[r];
    const int row = rowbase + r;
    if (row < SEQ) {
      #pragma unroll
      for (int c = 0; c < 4; c++)
        y[((size_t)b * SEQ + row) * CDIM + hh * 64 + c * 16 + col16] =
            f2bf(o[c][r] * inv);
    }
  }
}

// ---------------------------------------------------------------------------
extern "C" void kernel_launch(void* const* d_in, const int* in_sizes, int n_in,
                              void* d_out, int out_size, void* d_ws, size_t ws_size,
                              hipStream_t stream)
{
  const float* x  = (const float*)d_in[0];   // (B, L, C)
  const float* Wa = (const float*)d_in[1];   // (C, 3C)
  const float* Wp = (const float*)d_in[2];   // (C, C)
  float* out = (float*)d_out;                // (B, L, C) fp32

  // workspace carve-up (all element counts are multiples of 8 -> 16B aligned)
  u16* xb  = (u16*)d_ws;                                   // M x C bf16
  u16* wat = xb  + (size_t)MROWS * CDIM;                   // 3C x C bf16 (transposed)
  u16* wpt = wat + (size_t)N_QKV * CDIM;                   // C x C bf16 (transposed)
  u16* qb  = wpt + (size_t)CDIM * CDIM;                    // (B,H,L,d)
  u16* kb  = qb  + (size_t)BATCH * NH * SEQ * DH;          // (B,H,L,d)
  u16* vt  = kb  + (size_t)BATCH * NH * SEQ * DH;          // (B,H,d,SEQP) + slack
  u16* yb  = vt  + ((size_t)BATCH * NH * DH * SEQP + 64);  // (B,L,C)

  const size_t xtot = (size_t)MROWS * CDIM;
  cvt_x_kernel<<<dim3((unsigned)((xtot / 8 + 255) / 256)), dim3(256), 0, stream>>>(x, xb);
  transpose_w_kernel<<<dim3(N_QKV / 32, CDIM / 32), dim3(32, 8), 0, stream>>>(Wa, wat, CDIM, N_QKV);
  transpose_w_kernel<<<dim3(CDIM / 32, CDIM / 32), dim3(32, 8), 0, stream>>>(Wp, wpt, CDIM, CDIM);

  dim3 b256(256);
  gemm_kernel<0><<<dim3((MROWS + BM - 1) / BM, N_QKV / BN), b256, 0, stream>>>(
      xb, wat, qb, kb, vt, nullptr);

  attn_kernel<<<dim3(BATCH * NH * NQT), dim3(32), 0, stream>>>(qb, kb, vt, yb);

  gemm_kernel<1><<<dim3((MROWS + BM - 1) / BM, CDIM / BN), b256, 0, stream>>>(
      yb, wpt, nullptr, nullptr, nullptr, out);
}
